// MultiHeadSelfAttention_12945031430493
// MI455X (gfx1250) — compile-verified, matrix operand order checked by reading, not run on hardware
//
#include <hip/hip_runtime.h>
#include <hip/hip_bf16.h>

// ---------------------------------------------------------------------------
// MI455X (gfx1250) causal MHA + RoPE, bf16 WMMA pipeline.
//   proj<0>: Q = x @ w_q^T  -> RoPE -> [B,H,S,d]
//   proj<0>: K = x @ w_k^T  -> RoPE -> [B,H,S,d]
//   proj<1>: V = x @ w_v^T  ->         [B,H,d,S]   (transposed for PV B-operand)
//   attn   : flash-attention, v_wmma_f32_16x16x32_bf16 for QK^T and PV
//   proj<2>: out = attnO @ w_o^T ->    [B,S,DM]
// ---------------------------------------------------------------------------

#define S_LEN   2048
#define DMODEL  1024
#define NHEAD   16
#define HDIM    64
#define BATCH   4

typedef __attribute__((ext_vector_type(16))) __bf16 v16bf;
typedef __attribute__((ext_vector_type(8)))  __bf16 v8bf;
typedef __attribute__((ext_vector_type(8)))  float  v8f;

struct FragBF { union { v16bf v; __bf16 e[16]; }; };
struct FragF  { union { v8f   v; float  e[8];  }; };

__device__ __forceinline__ v8f wmma_bf16(v16bf a, v16bf b, v8f c) {
    // v_wmma_f32_16x16x32_bf16  (neg_a, A, neg_b, B, c_mod, C, reuse_a, reuse_b)
    return __builtin_amdgcn_wmma_f32_16x16x32_bf16(false, a, false, b, (short)0, c,
                                                   false, false);
}

// pack 8 consecutive fp32 (two float4) into fragment positions [base..base+7]
__device__ __forceinline__ void pack8(FragBF& f, int base, float4 a, float4 b) {
    f.e[base + 0] = (__bf16)a.x; f.e[base + 1] = (__bf16)a.y;
    f.e[base + 2] = (__bf16)a.z; f.e[base + 3] = (__bf16)a.w;
    f.e[base + 4] = (__bf16)b.x; f.e[base + 5] = (__bf16)b.y;
    f.e[base + 6] = (__bf16)b.z; f.e[base + 7] = (__bf16)b.w;
}

// Load a 16x32 bf16 fragment row from fp32 global memory (K-contiguous rows).
__device__ __forceinline__ void ld_frag_global(FragBF& f, const float* rowp, int klo) {
    const float4 c0 = *(const float4*)(rowp + klo);
    const float4 c1 = *(const float4*)(rowp + klo + 4);
    const float4 c2 = *(const float4*)(rowp + klo + 16);
    const float4 c3 = *(const float4*)(rowp + klo + 20);
    pack8(f, 0, c0, c1);
    pack8(f, 8, c2, c3);
}

// Load a 16x32 bf16 A/B fragment from LDS row-major (stride 48 bf16 = 96B).
// ISA layout: lanes 0-15 hold K {0..7,16..23}, lanes 16-31 hold K {8..15,24..31}.
__device__ __forceinline__ void ld_frag_lds(FragBF& f, const __bf16* rowp, int klo) {
    const v8bf c0 = *(const v8bf*)(rowp + klo);
    const v8bf c1 = *(const v8bf*)(rowp + 16 + klo);
#pragma unroll
    for (int i = 0; i < 8; ++i) { f.e[i] = c0[i]; f.e[8 + i] = c1[i]; }
}

// ---------------------------------------------------------------------------
// Generic projection GEMM: Y[row,e] = sum_m X[row,m] * W[e,m]
// M = B*S = 8192, N = K = 1024.  Block tile 256(M) x 64(N); 8 waves x (32x64).
// Each B fragment read feeds 2 WMMAs (B reuse across the 2 M-subtiles).
// MODE 0: write [B,H,S,d] with RoPE.  MODE 1: write [B,H,d,S].  MODE 2: [row,e].
// ---------------------------------------------------------------------------
template <int MODE>
__global__ __launch_bounds__(256)
void proj_kernel(const float* __restrict__ X, const float* __restrict__ W,
                 const int* __restrict__ pos, float* __restrict__ Y)
{
    __shared__ __bf16 As[256 * 48];     // 24 KB
    __shared__ __bf16 Bs[64 * 48];      //  6 KB

    const int tid  = threadIdx.x;
    const int lane = tid & 31;
    const int wv   = tid >> 5;                 // 0..7 waves
    const int m0   = blockIdx.x * 256;
    const int n0   = blockIdx.y * 64;
    const int ln   = lane & 15;
    const int hf   = lane >> 4;
    const int klo  = hf * 8;

    FragF c[2][4];                             // [M-subtile][N-subtile]
#pragma unroll
    for (int p = 0; p < 2; ++p)
#pragma unroll
        for (int s = 0; s < 4; ++s)
#pragma unroll
            for (int r = 0; r < 8; ++r) c[p][s].e[r] = 0.0f;

    for (int k0 = 0; k0 < DMODEL; k0 += 32) {
        // stage A tile 256x32 (2048 float4, 8 per thread), fp32 -> bf16
#pragma unroll
        for (int i = 0; i < 8; ++i) {
            const int idx = tid + i * 256;
            const int row = idx >> 3, fq = idx & 7;
            const float4 v = *(const float4*)(X + (size_t)(m0 + row) * DMODEL + k0 + fq * 4);
            __bf16* d = &As[row * 48 + fq * 4];
            d[0] = (__bf16)v.x; d[1] = (__bf16)v.y; d[2] = (__bf16)v.z; d[3] = (__bf16)v.w;
        }
        // stage B tile 64x32 (512 float4, 2 per thread)
#pragma unroll
        for (int i = 0; i < 2; ++i) {
            const int idx = tid + i * 256;
            const int row = idx >> 3, fq = idx & 7;
            const float4 v = *(const float4*)(W + (size_t)(n0 + row) * DMODEL + k0 + fq * 4);
            __bf16* d = &Bs[row * 48 + fq * 4];
            d[0] = (__bf16)v.x; d[1] = (__bf16)v.y; d[2] = (__bf16)v.z; d[3] = (__bf16)v.w;
        }
        if (k0 + 32 < DMODEL)   // speculative prefetch of next A tile (global_prefetch_b8)
            __builtin_prefetch(X + (size_t)(m0 + tid) * DMODEL + k0 + 32, 0, 1);
        __syncthreads();

        FragBF a[2];
        ld_frag_lds(a[0], &As[(wv * 32 + ln) * 48], klo);
        ld_frag_lds(a[1], &As[(wv * 32 + 16 + ln) * 48], klo);
#pragma unroll
        for (int sub = 0; sub < 4; ++sub) {
            FragBF b;
            ld_frag_lds(b, &Bs[(sub * 16 + ln) * 48], klo);
            c[0][sub].v = wmma_bf16(a[0].v, b.v, c[0][sub].v);
            c[1][sub].v = wmma_bf16(a[1].v, b.v, c[1][sub].v);
        }
        __syncthreads();
    }

    // epilogue
#pragma unroll
    for (int p = 0; p < 2; ++p) {
#pragma unroll
        for (int sub = 0; sub < 4; ++sub) {
            const int e = n0 + sub * 16 + ln;          // output feature
#pragma unroll
            for (int r = 0; r < 8; ++r) {
                const int row = m0 + wv * 32 + p * 16 + r + hf * 8;  // global M row
                const int b   = row >> 11;                           // / 2048
                const int s   = row & (S_LEN - 1);
                float val = c[p][sub].e[r];
                if (MODE == 0) {
                    const int h = e >> 6, dc = e & 63;
                    const float partner = __shfl_xor(val, 1);        // adjacent head-dim col
                    const int   i2   = dc >> 1;
                    const float freq = __expf((float)i2 * (-2.0f / 64.0f) * 9.210340371976184f);
                    const float ang  = (float)pos[s] * freq;
                    float sn, cs; __sincosf(ang, &sn, &cs);
                    val = ((dc & 1) == 0) ? (cs * val - sn * partner)
                                          : (sn * partner + cs * val);
                    Y[(((size_t)(b * NHEAD + h)) * S_LEN + s) * HDIM + dc] = val;
                } else if (MODE == 1) {
                    const int h = e >> 6, dc = e & 63;
                    Y[(((size_t)(b * NHEAD + h)) * HDIM + dc) * S_LEN + s] = val;
                } else {
                    Y[(size_t)row * DMODEL + e] = val;
                }
            }
        }
    }
}

// ---------------------------------------------------------------------------
// Flash attention: one wave per 32-query tile (2 x 16-row WMMA tiles sharing
// the K/V fragment loads), key blocks of 32.
// Q/K: [B,H,S,d]; VT: [B,H,d,S]; O: [B,S,DM].
// ---------------------------------------------------------------------------
__global__ __launch_bounds__(128)
void attn_kernel(const float* __restrict__ Q, const float* __restrict__ K,
                 const float* __restrict__ VT, float* __restrict__ O)
{
    __shared__ __bf16 pb[4][16 * 48];          // per-wave P staging (C-layout -> A-layout)

    const int lane = threadIdx.x & 31;
    const int wv   = threadIdx.x >> 5;
    const int bh   = blockIdx.y;               // b*16 + h
    const int q0   = (blockIdx.x * 4 + wv) * 32;
    const int ln   = lane & 15;
    const int hf   = lane >> 4;
    const int klo  = hf * 8;
    const size_t base = (size_t)bh * S_LEN * HDIM;   // also valid for VT ([bh,64,2048])

    // Q fragments: 2 q-subtiles x 2 K-halves, direct from global.
    FragBF qa[2][2];
#pragma unroll
    for (int qt = 0; qt < 2; ++qt) {
        const float* qrow = Q + base + (size_t)(q0 + qt * 16 + ln) * HDIM;
#pragma unroll
        for (int u = 0; u < 2; ++u)
            ld_frag_global(qa[qt][u], qrow + u * 32, klo);
    }

    float mstat[2][8], lstat[2][8];
    FragF o[2][4];
#pragma unroll
    for (int qt = 0; qt < 2; ++qt) {
#pragma unroll
        for (int r = 0; r < 8; ++r) { mstat[qt][r] = -1e30f; lstat[qt][r] = 0.0f; }
#pragma unroll
        for (int d = 0; d < 4; ++d)
#pragma unroll
            for (int r = 0; r < 8; ++r) o[qt][d].e[r] = 0.0f;
    }

    for (int j = 0; j <= q0 + 31; j += 32) {
        // ---- S = Q K^T: K fragments loaded once, reused by both q-subtiles ----
        FragF st[2][2];                        // [q-subtile][key-subtile]
#pragma unroll
        for (int t = 0; t < 2; ++t) {
            const float* krow = K + base + (size_t)(j + t * 16 + ln) * HDIM;
            FragBF kb[2];
            ld_frag_global(kb[0], krow, klo);
            ld_frag_global(kb[1], krow + 32, klo);
#pragma unroll
            for (int qt = 0; qt < 2; ++qt) {
                FragF acc;
#pragma unroll
                for (int r = 0; r < 8; ++r) acc.e[r] = 0.0f;
                acc.v = wmma_bf16(qa[qt][0].v, kb[0].v, acc.v);
                acc.v = wmma_bf16(qa[qt][1].v, kb[1].v, acc.v);
                st[qt][t] = acc;
            }
        }

        // ---- V fragments: loaded once, reused by both q-subtiles ----
        FragBF vb[4];
#pragma unroll
        for (int dsub = 0; dsub < 4; ++dsub) {
            const float* vrow = VT + base + (size_t)(dsub * 16 + ln) * S_LEN + j;
            ld_frag_global(vb[dsub], vrow, klo);
        }

        // ---- per q-subtile: causal mask + online softmax + PV ----
#pragma unroll
        for (int qt = 0; qt < 2; ++qt) {
#pragma unroll
            for (int r = 0; r < 8; ++r) {
                const int qg = q0 + qt * 16 + r + hf * 8;
#pragma unroll
                for (int t = 0; t < 2; ++t) {
                    const int kg = j + t * 16 + ln;
                    const float sv = st[qt][t].e[r] * 0.125f;    // 1/sqrt(64)
                    st[qt][t].e[r] = (kg > qg) ? -1e30f : sv;
                }
                float mx = fmaxf(st[qt][0].e[r], st[qt][1].e[r]);
#pragma unroll
                for (int m = 1; m < 16; m <<= 1) mx = fmaxf(mx, __shfl_xor(mx, m));
                const float mnew  = fmaxf(mstat[qt][r], mx);
                const float alpha = __expf(mstat[qt][r] - mnew);
                mstat[qt][r] = mnew;
                const float p0 = __expf(st[qt][0].e[r] - mnew);
                const float p1 = __expf(st[qt][1].e[r] - mnew);
                st[qt][0].e[r] = p0; st[qt][1].e[r] = p1;
                float rs = p0 + p1;
#pragma unroll
                for (int m = 1; m < 16; m <<= 1) rs += __shfl_xor(rs, m);
                lstat[qt][r] = lstat[qt][r] * alpha + rs;
#pragma unroll
                for (int d = 0; d < 4; ++d) o[qt][d].e[r] *= alpha;
            }

            // P (C-layout) -> LDS -> A fragment (in-order DS within a wave)
            __bf16* pw = pb[wv];
#pragma unroll
            for (int r = 0; r < 8; ++r) {
                const int row = r + hf * 8;
                pw[row * 48 + ln]      = (__bf16)st[qt][0].e[r];
                pw[row * 48 + 16 + ln] = (__bf16)st[qt][1].e[r];
            }
            FragBF pa;
            ld_frag_lds(pa, pw + ln * 48, klo);

#pragma unroll
            for (int dsub = 0; dsub < 4; ++dsub)
                o[qt][dsub].v = wmma_bf16(pa.v, vb[dsub].v, o[qt][dsub].v);
        }
    }

    // ---- normalize + write [B,S,DM] ----
    const int b = bh >> 4, h = bh & 15;
#pragma unroll
    for (int qt = 0; qt < 2; ++qt) {
#pragma unroll
        for (int dsub = 0; dsub < 4; ++dsub) {
            const int col = h * HDIM + dsub * 16 + ln;
#pragma unroll
            for (int r = 0; r < 8; ++r) {
                const int sg = q0 + qt * 16 + r + hf * 8;
                O[((size_t)b * S_LEN + sg) * DMODEL + col] = o[qt][dsub].e[r] / lstat[qt][r];
            }
        }
    }
}

// ---------------------------------------------------------------------------
extern "C" void kernel_launch(void* const* d_in, const int* in_sizes, int n_in,
                              void* d_out, int out_size, void* d_ws, size_t ws_size,
                              hipStream_t stream)
{
    (void)in_sizes; (void)n_in; (void)out_size; (void)ws_size;
    const float* x   = (const float*)d_in[0];
    const float* w_q = (const float*)d_in[1];
    const float* w_k = (const float*)d_in[2];
    const float* w_v = (const float*)d_in[3];
    const float* w_o = (const float*)d_in[4];
    const int*   pos = (const int*)d_in[5];
    float*       out = (float*)d_out;

    const size_t T = (size_t)BATCH * NHEAD * S_LEN * HDIM;   // 8.4M floats each
    float* Qb = (float*)d_ws;       // [B,H,S,d]
    float* Kb = Qb + T;             // [B,H,S,d]
    float* Vt = Kb + T;             // [B,H,d,S]
    float* AO = Vt + T;             // [B,S,DM]

    const dim3 ggrid(BATCH * S_LEN / 256, DMODEL / 64);      // (32,16)
    const dim3 gblock(256);
    proj_kernel<0><<<ggrid, gblock, 0, stream>>>(x,  w_q, pos, Qb);
    proj_kernel<0><<<ggrid, gblock, 0, stream>>>(x,  w_k, pos, Kb);
    proj_kernel<1><<<ggrid, gblock, 0, stream>>>(x,  w_v, pos, Vt);

    // 32 queries per wave, 4 waves per block: grid.x = 2048 / 128 = 16
    attn_kernel<<<dim3(S_LEN / 128, BATCH * NHEAD), 128, 0, stream>>>(Qb, Kb, Vt, AO);

    proj_kernel<2><<<ggrid, gblock, 0, stream>>>(AO, w_o, pos, out);
}
